// BiLSTM_CRF_20572893347919
// MI455X (gfx1250) — matrix operand order repaired and checked by reading, not compiled, not run on hardware
//
#include <hip/hip_runtime.h>
#include <math.h>

#define Bn 128
#define Tn 512
#define Dn 256
#define Hn 128
#define G4 512   // 4*H
#define NROW ((size_t)Tn * Bn)   // 65536 flattened (t,b) rows

typedef __attribute__((ext_vector_type(8)))  unsigned short v8us;
typedef __attribute__((ext_vector_type(16))) unsigned short v16us;
typedef __attribute__((ext_vector_type(16))) __bf16         v16bf;
typedef __attribute__((ext_vector_type(8)))  float          v8f;

__device__ __forceinline__ unsigned short f2bf(float f){
    unsigned u = __builtin_bit_cast(unsigned, f);
    u += 0x7FFFu + ((u >> 16) & 1u);          // round-nearest-even
    return (unsigned short)(u >> 16);
}
__device__ __forceinline__ float bf2f(unsigned short h){
    unsigned u = ((unsigned)h) << 16;
    return __builtin_bit_cast(float, u);
}
__device__ __forceinline__ float sigm(float x){ return 1.0f / (1.0f + __expf(-x)); }

// A-fragment (16x32 bf16, M x K): lane m=L%16, h=L/16; elems 0..7 at k0+h*8, 8..15 at k0+16+h*8
__device__ __forceinline__ v16bf load_frag_split(const unsigned short* p /* row + k0 */, int h){
    v8us lo = *(const v8us*)(p + h * 8);
    v8us hi = *(const v8us*)(p + 16 + h * 8);
    v16us a = __builtin_shufflevector(lo, hi, 0,1,2,3,4,5,6,7,8,9,10,11,12,13,14,15);
    return __builtin_bit_cast(v16bf, a);
}
// B-fragment (32x16 bf16, K x N): lane n=L%16, h=L/16; elems 0..7 at k0+h*8, 8..15 at k0+16+h*8
__device__ __forceinline__ v16bf load_frag_cont(const unsigned short* p /* row + k0 */, int h){
    v8us lo = *(const v8us*)(p + h * 8);
    v8us hi = *(const v8us*)(p + 16 + h * 8);
    v16us b = __builtin_shufflevector(lo, hi, 0,1,2,3,4,5,6,7,8,9,10,11,12,13,14,15);
    return __builtin_bit_cast(v16bf, b);
}

// Tiled G layout: row-tile mt = row/16 (rows are t*B+b); element (m=8h+j, n) of tile mt
// lives at bf16 index ((mt*512 + n)*2 + h)*8 + j  -> each lane's 8 C-values are 16B contiguous.
__device__ __forceinline__ size_t g_tile_off(size_t mt, int n, int h){
    return ((mt * G4 + (size_t)n) * 2 + (size_t)h) * 8;
}

// ---------------- K0: embedding gather + f32 -> bf16 (vectorized) ----------------
__global__ void k_embed(const int* __restrict__ pad_seq, const float* __restrict__ emb,
                        unsigned short* __restrict__ X){
    int tid = threadIdx.x;
    int r = blockIdx.x * 4 + (tid >> 6);     // r = t*B + b
    int t = r >> 7, b = r & 127;
    int tok = pad_seq[b * Tn + t];
    int d = (tid & 63) * 4;
    const float4 v = *(const float4*)(emb + (size_t)tok * Dn + d);
    ushort4 o;
    o.x = f2bf(v.x); o.y = f2bf(v.y); o.z = f2bf(v.z); o.w = f2bf(v.w);
    *(ushort4*)(X + (size_t)r * Dn + d) = o;
}

// ---------------- K1: weights f32 -> bf16 [Wihf|Wihb|Whhf|Whhb] ----------------
__global__ void k_wconv(const float* __restrict__ Wihf, const float* __restrict__ Wihb,
                        const float* __restrict__ Whhf, const float* __restrict__ Whhb,
                        unsigned short* __restrict__ Wbf){
    int id = blockIdx.x * 256 + threadIdx.x;    // 0..393215
    float v;
    if      (id < 131072) v = Wihf[id];
    else if (id < 262144) v = Wihb[id - 131072];
    else if (id < 327680) v = Whhf[id - 262144];
    else                  v = Whhb[id - 327680];
    Wbf[id] = f2bf(v);
}

// ---------------- K2: G = X @ Wih^T   (bf16 in, f32 acc, bf16 tiled out) ----------------
// block: 256 thr = 8 waves; wave owns 16 M-rows; block tile = 128M x 64N; grid=(16,512)
__global__ void __launch_bounds__(256) k_gemm_in(const unsigned short* __restrict__ X,
                          const unsigned short* __restrict__ Wbf,
                          unsigned short* __restrict__ Gf,
                          unsigned short* __restrict__ Gb){
    int tid  = threadIdx.x;
    int wave = tid >> 5, lane = tid & 31;
    int h = lane >> 4, ln = lane & 15;
    int bx = blockIdx.x;
    int dir = bx >> 3;
    int n0base = (bx & 7) * 64;
    const unsigned short* Wih = Wbf + (size_t)dir * (G4 * Dn);
    unsigned short* G = dir ? Gb : Gf;
    int Mw = blockIdx.y * 128 + wave * 16;
    size_t mt = (size_t)(Mw >> 4);            // global 16-row tile index

    v8f acc[4];
    #pragma unroll
    for (int nt = 0; nt < 4; ++nt)
        #pragma unroll
        for (int jj = 0; jj < 8; ++jj) acc[nt][jj] = 0.0f;

    const unsigned short* Arow = X + (size_t)(Mw + ln) * Dn;
    #pragma unroll
    for (int kk = 0; kk < 8; ++kk){                     // K = 256 in steps of 32
        v16bf af = load_frag_split(Arow + kk * 32, h);
        #pragma unroll
        for (int nt = 0; nt < 4; ++nt){
            const unsigned short* Brow = Wih + (size_t)(n0base + nt * 16 + ln) * Dn + kk * 32;
            v16bf bfr = load_frag_cont(Brow, h);
            acc[nt] = __builtin_amdgcn_wmma_f32_16x16x32_bf16(
                false, af, false, bfr, (short)0, acc[nt], false, false);
        }
    }
    #pragma unroll
    for (int nt = 0; nt < 4; ++nt){
        int ncol = n0base + nt * 16 + ln;
        v8us o;
        #pragma unroll
        for (int jj = 0; jj < 8; ++jj) o[jj] = f2bf(acc[nt][jj]);
        *(v8us*)(G + g_tile_off(mt, ncol, h)) = o;       // one 16B store per tile
    }
}

// ---------------- K3: persistent recurrence. grid=(8 batch-tiles, 2 dirs), 256 thr ----------------
__global__ void __launch_bounds__(256) k_lstm(const unsigned short* __restrict__ Gf,
                      const unsigned short* __restrict__ Gb,
                      const unsigned short* __restrict__ Whhf_bf,
                      const unsigned short* __restrict__ Whhb_bf,
                      const float* __restrict__ bihf, const float* __restrict__ bhhf,
                      const float* __restrict__ bihb, const float* __restrict__ bhhb,
                      const int* __restrict__ lens,
                      unsigned short* __restrict__ hf,
                      unsigned short* __restrict__ hb){
    __shared__ float gates[16][G4];          // 32 KB: pre-activations for this step
    __shared__ unsigned short hst[16 * Hn];  // 4 KB: h state (A-fragments), bf16

    int tid = threadIdx.x;
    int dir = blockIdx.y;
    int b0  = blockIdx.x * 16;
    const unsigned short* Gd  = dir ? Gb : Gf;
    const unsigned short* Whh = dir ? Whhb_bf : Whhf_bf;
    const float* bih = dir ? bihb : bihf;
    const float* bhh = dir ? bhhb : bhhf;
    unsigned short* hout = dir ? hb : hf;

    int wave = tid >> 5, lane = tid & 31, h = lane >> 4, ln = lane & 15;
    int colbase = wave * 64;                 // each of 8 waves owns 64 gate columns

    // --- hoist ALL Whh B-fragments into registers (16 frags = 128 VGPRs, loaded once) ---
    v16bf bfrag[16];
    #pragma unroll
    for (int kk = 0; kk < 4; ++kk)
        #pragma unroll
        for (int nt = 0; nt < 4; ++nt){
            int nrow = colbase + nt * 16 + ln;
            bfrag[kk * 4 + nt] = load_frag_cont(Whh + (size_t)nrow * Hn + kk * 32, h);
        }

    // activation-phase mapping: thread owns column jc for 8 batch rows
    int jc = tid & 127;
    int rb = (tid >> 7) * 8;
    float bias_i = bih[jc]          + bhh[jc];
    float bias_f = bih[Hn + jc]     + bhh[Hn + jc];
    float bias_g = bih[2 * Hn + jc] + bhh[2 * Hn + jc];
    float bias_o = bih[3 * Hn + jc] + bhh[3 * Hn + jc];
    float creg[8];
    int   lenr[8];
    #pragma unroll
    for (int rr = 0; rr < 8; ++rr){ creg[rr] = 0.0f; lenr[rr] = lens[b0 + rb + rr]; }

    for (int i = tid; i < 16 * Hn; i += 256) hst[i] = 0;
    __syncthreads();

    for (int step = 0; step < Tn; ++step){
        int tt = dir ? (Tn - 1 - step) : step;
        size_t mt = (size_t)tt * (Bn / 16) + (b0 >> 4);   // this WG's G tile at time tt

        // prefetch next step's contiguous 16 KB gate tile (hide HBM latency behind WMMA)
        if (step + 1 < Tn){
            int tn = dir ? (tt - 1) : (tt + 1);
            size_t mtn = (size_t)tn * (Bn / 16) + (b0 >> 4);
            const unsigned short* pp = Gd + mtn * (G4 * 16) + tid * 32;
            __builtin_prefetch(pp, 0, 1);
        }

        // init accumulators from precomputed input projection: one 16B load per tile
        v8f acc[4];
        #pragma unroll
        for (int nt = 0; nt < 4; ++nt){
            int ncol = colbase + nt * 16 + ln;
            v8us gv = *(const v8us*)(Gd + g_tile_off(mt, ncol, h));
            #pragma unroll
            for (int jj = 0; jj < 8; ++jj) acc[nt][jj] = bf2f(gv[jj]);
        }
        // h @ Whh^T : M=16, N=64/wave, K=128, register-resident B
        #pragma unroll
        for (int kk = 0; kk < 4; ++kk){
            v16bf af = load_frag_split(&hst[ln * Hn + kk * 32], h);
            #pragma unroll
            for (int nt = 0; nt < 4; ++nt){
                acc[nt] = __builtin_amdgcn_wmma_f32_16x16x32_bf16(
                    false, af, false, bfrag[kk * 4 + nt], (short)0, acc[nt], false, false);
            }
        }
        #pragma unroll
        for (int nt = 0; nt < 4; ++nt){
            int ncol = colbase + nt * 16 + ln;
            #pragma unroll
            for (int jj = 0; jj < 8; ++jj) gates[8 * h + jj][ncol] = acc[nt][jj];
        }
        __syncthreads();

        // gate nonlinearities + state update (masked pack semantics)
        #pragma unroll
        for (int rr = 0; rr < 8; ++rr){
            int r = rb + rr;
            float gi  = gates[r][jc]          + bias_i;
            float gfv = gates[r][Hn + jc]     + bias_f;
            float gg  = gates[r][2 * Hn + jc] + bias_g;
            float go  = gates[r][3 * Hn + jc] + bias_o;
            float c   = creg[rr];
            float c2  = sigm(gfv) * c + sigm(gi) * tanhf(gg);
            float h2  = sigm(go) * tanhf(c2);
            bool msk  = (tt < lenr[rr]);
            if (msk){ creg[rr] = c2; hst[r * Hn + jc] = f2bf(h2); }
            hout[((size_t)tt * Bn + b0 + r) * Hn + jc] = msk ? f2bf(h2) : (unsigned short)0;
        }
        __syncthreads();
    }
}

// ---------------- K4: emissions  scores[t,b,l] = [hf|hb] . Wlab[l] + blab[l] ----------------
__global__ void k_emit(const unsigned short* __restrict__ hf, const unsigned short* __restrict__ hb,
                       const float* __restrict__ Wlab, const float* __restrict__ blab,
                       float* __restrict__ scores){
    int id = blockIdx.x * 256 + threadIdx.x;       // (t*B + b)
    const unsigned short* pf = hf + (size_t)id * Hn;
    const unsigned short* pb = hb + (size_t)id * Hn;
    float s[4] = { blab[0], blab[1], blab[2], blab[3] };
    for (int d0 = 0; d0 < Hn; d0 += 8){
        v8us cf = *(const v8us*)(pf + d0);
        v8us cb = *(const v8us*)(pb + d0);
        #pragma unroll
        for (int e = 0; e < 8; ++e){
            float xf = bf2f(cf[e]);
            float xb = bf2f(cb[e]);
            #pragma unroll
            for (int l = 0; l < 4; ++l)
                s[l] += xf * Wlab[l * Dn + d0 + e] + xb * Wlab[l * Dn + Hn + d0 + e];
        }
    }
    #pragma unroll
    for (int l = 0; l < 4; ++l) scores[(size_t)id * 4 + l] = s[l];
}

// ---------------- K5: Viterbi (1 thread per batch row) ----------------
__global__ void k_viterbi(const float* __restrict__ scores,
                          const float* __restrict__ trans,
                          const float* __restrict__ fromB,
                          const float* __restrict__ toE,
                          const int* __restrict__ lens,
                          unsigned* __restrict__ btb,
                          float* __restrict__ out){
    int b = threadIdx.x;
    if (b >= Bn) return;
    int len = lens[b];
    float tr[16];
    #pragma unroll
    for (int i = 0; i < 16; ++i) tr[i] = trans[i];
    float best[4];
    #pragma unroll
    for (int l = 0; l < 4; ++l) best[l] = fromB[l] + scores[(size_t)b * 4 + l];

    for (int ti = 1; ti < Tn; ++ti){
        if (ti >= len) break;                    // best frozen past len; bt never read there
        float e[4];
        #pragma unroll
        for (int l = 0; l < 4; ++l) e[l] = scores[((size_t)ti * Bn + b) * 4 + l];
        float nb[4]; unsigned pack = 0u;
        #pragma unroll
        for (int cur = 0; cur < 4; ++cur){
            float mx = best[0] + tr[cur];
            int   arg = 0;
            #pragma unroll
            for (int p = 1; p < 4; ++p){
                float v = best[p] + tr[p * 4 + cur];
                if (v > mx){ mx = v; arg = p; }  // first-max semantics like jnp.argmax
            }
            nb[cur] = mx + e[cur];
            pack |= ((unsigned)arg) << (8 * cur);
        }
        btb[(size_t)ti * Bn + b] = pack;
        #pragma unroll
        for (int l = 0; l < 4; ++l) best[l] = nb[l];
    }
    int   last = 0;
    float mx = best[0] + toE[0];
    for (int l = 1; l < 4; ++l){ float v = best[l] + toE[l]; if (v > mx){ mx = v; last = l; } }

    int cur = 0;
    for (int ti = Tn - 1; ti >= 0; --ti){
        if (ti == len - 1) cur = last;
        else if (ti < len - 1){
            unsigned p = btb[(size_t)(ti + 1) * Bn + b];
            cur = (int)((p >> (8 * cur)) & 0xFFu);
        }
        out[(size_t)b * Tn + ti] = (ti < len) ? (float)cur : 0.0f;
    }
}

// ---------------- launcher ----------------
extern "C" void kernel_launch(void* const* d_in, const int* in_sizes, int n_in,
                              void* d_out, int out_size, void* d_ws, size_t ws_size,
                              hipStream_t stream){
    (void)in_sizes; (void)n_in; (void)out_size; (void)ws_size;
    const int*   pad_seq = (const int*)  d_in[0];
    const int*   lens    = (const int*)  d_in[1];
    const float* emb     = (const float*)d_in[2];
    const float* Wihf    = (const float*)d_in[3];
    const float* Whhf    = (const float*)d_in[4];
    const float* bihf    = (const float*)d_in[5];
    const float* bhhf    = (const float*)d_in[6];
    const float* Wihb    = (const float*)d_in[7];
    const float* Whhb    = (const float*)d_in[8];
    const float* bihb    = (const float*)d_in[9];
    const float* bhhb    = (const float*)d_in[10];
    const float* Wlab    = (const float*)d_in[11];
    const float* blab    = (const float*)d_in[12];
    const float* trans   = (const float*)d_in[13];
    const float* fromB   = (const float*)d_in[14];
    const float* toE     = (const float*)d_in[15];

    char* ws = (char*)d_ws;
    const size_t OFF_X  = 0;                                  // X bf16: 65536*256*2 = 33,554,432
    const size_t OFF_W  = OFF_X + NROW * Dn * 2;              // weights bf16: 786,432
    const size_t W_ELEM = (size_t)2 * G4 * Dn + (size_t)2 * G4 * Hn;   // 393216
    const size_t OFF_GF = OFF_W + W_ELEM * 2;                 // Gf bf16 (tiled): 67,108,864
    const size_t OFF_GB = OFF_GF + NROW * G4 * 2;             // Gb bf16 (tiled): 67,108,864

    unsigned short* X   = (unsigned short*)(ws + OFF_X);
    unsigned short* Wbf = (unsigned short*)(ws + OFF_W);
    unsigned short* Gf  = (unsigned short*)(ws + OFF_GF);
    unsigned short* Gb  = (unsigned short*)(ws + OFF_GB);
    // aliases: X dead after GEMM -> reuse for h outputs; Gf dead after recurrence -> scores/backtrace
    unsigned short* hf  = (unsigned short*)(ws + OFF_X);
    unsigned short* hb  = hf + NROW * Hn;
    float*    scores    = (float*)(ws + OFF_GF);
    unsigned* btb       = (unsigned*)(ws + OFF_GF + NROW * 4 * sizeof(float));

    k_embed  <<<(Tn * Bn) / 4, 256, 0, stream>>>(pad_seq, emb, X);
    k_wconv  <<<1536, 256, 0, stream>>>(Wihf, Wihb, Whhf, Whhb, Wbf);
    k_gemm_in<<<dim3(16, 512), 256, 0, stream>>>(X, Wbf, Gf, Gb);
    k_lstm   <<<dim3(8, 2), 256, 0, stream>>>(Gf, Gb,
                 Wbf + (size_t)2 * G4 * Dn,                       // Whhf bf16
                 Wbf + (size_t)2 * G4 * Dn + (size_t)G4 * Hn,     // Whhb bf16
                 bihf, bhhf, bihb, bhhb, lens, hf, hb);
    k_emit   <<<256, 256, 0, stream>>>(hf, hb, Wlab, blab, scores);
    k_viterbi<<<1, 128, 0, stream>>>(scores, trans, fromB, toE, lens, btb, (float*)d_out);
}
